// DeepSeekMoE_34720515620990
// MI455X (gfx1250) — matrix operand order repaired and checked by reading, not compiled, not run on hardware
//
#include <hip/hip_runtime.h>

// ---------------------------------------------------------------------------
// DeepSeek-style MoE for MI455X (gfx1250, wave32, WMMA bf16).
// Memory-bound (~500MB min traffic @ 23.3 TB/s ~= 21us). All GEMMs use
// v_wmma_f32_16x16x32_bf16, 64x64 block tile (8 waves x two 16x16 tiles
// sharing one B fragment). Weights are pre-converted ONCE to bf16 and
// TRANSPOSED to [N][K] so both A and B fragments are contiguous 16B
// global_load_b128 -- no per-element conversion in the hot loop.
// ---------------------------------------------------------------------------

typedef __bf16 bf16_t;
typedef __attribute__((ext_vector_type(16))) __bf16       v16bf;
typedef __attribute__((ext_vector_type(8)))  float        v8f;
typedef __attribute__((ext_vector_type(4)))  unsigned int u32x4;

constexpr int D    = 2048;
constexpr int E    = 16;
constexpr int H_R  = 512;
constexpr int H_S  = 8192;
constexpr int NTOK = 4 * 2048;      // 8192 tokens
constexpr int NA   = NTOK * 2;      // 16384 routed assignments (top-2)
constexpr int CH   = 2048;          // shared-expert hidden chunk
constexpr int NCH  = H_S / CH;      // 4 chunks

#define DEV __device__ __forceinline__

DEV bf16_t f2bf(float f) {
    union { float f; unsigned u; } in; in.f = f;
    unsigned r = in.u + 0x7FFFu + ((in.u >> 16) & 1u);   // round-to-nearest-even
    union { unsigned short s; bf16_t b; } out;
    out.s = (unsigned short)(r >> 16);
    return out.b;
}

DEV float gelu_exact(float v) {
    return 0.5f * v * (1.0f + erff(v * 0.70710678118654752440f));
}

// A fragment (16x32 bf16, this lane's row): K = kbase+0..7 and kbase+16..23,
// i.e. two contiguous 16-byte runs (kbase = (lane>>4)*8 handled by caller).
DEV v16bf load_a_frag(const bf16_t* row, int koff) {
    union { u32x4 u[2]; v16bf v; } c;
    c.u[0] = *(const u32x4*)(row + koff);
    c.u[1] = *(const u32x4*)(row + koff + 16);
    return c.v;
}

// B fragment from TRANSPOSED bf16 weights (Bt layout [N][K]): lane holds
// column n, values K = kb16+0..7 and kb16+16..23 -> two 16-byte runs.
DEV v16bf load_bt_frag(const bf16_t* col, int koff) {
    union { u32x4 u[2]; v16bf v; } c;
    c.u[0] = *(const u32x4*)(col + koff);
    c.u[1] = *(const u32x4*)(col + koff + 16);
    return c.v;
}

// Two 16x16 tiles of one column, 32 rows apart, sharing one B fragment, software
// pipelined so next-k loads issue before this-k WMMAs.
DEV void wave_gemm2(const bf16_t* a0row, const bf16_t* a1row,
                    const bf16_t* bcol, int K, v8f& acc0, v8f& acc1)
{
    const int lane = threadIdx.x & 31;
    const int kbA  = (lane >> 4) * 8;
    const int kb16 = (lane >> 4) * 8;
    v16bf b  = load_bt_frag(bcol, kb16);
    v16bf a0 = load_a_frag(a0row, kbA);
    v16bf a1 = load_a_frag(a1row, kbA);
    for (int k0 = 0; k0 < K; k0 += 32) {
        const int kn = (k0 + 32 < K) ? (k0 + 32) : k0;   // last iter: reload self
        v16bf bn  = load_bt_frag(bcol + kn, kb16);
        v16bf an0 = load_a_frag(a0row + kn, kbA);
        v16bf an1 = load_a_frag(a1row + kn, kbA);
        acc0 = __builtin_amdgcn_wmma_f32_16x16x32_bf16(false, a0, false, b,
                                                       (short)0, acc0, false, false);
        acc1 = __builtin_amdgcn_wmma_f32_16x16x32_bf16(false, a1, false, b,
                                                       (short)0, acc1, false, false);
        b = bn; a0 = an0; a1 = an1;
    }
}

// ---------------------------------------------------------------------------
// Weight pre-pass: f32 [R][C] -> bf16 [C][R] (transpose + convert), one
// matrix per blockIdx.z. 32x32 tile via LDS, fully coalesced both sides.
// ---------------------------------------------------------------------------
__global__ void __launch_bounds__(256)
k_transpose_cvt(const float* __restrict__ src, bf16_t* __restrict__ dst,
                int R, int C)
{
    src += (size_t)blockIdx.z * R * C;
    dst += (size_t)blockIdx.z * R * C;
    __shared__ float tile[32][33];
    const int c0 = blockIdx.x * 32, r0 = blockIdx.y * 32;
    const int tc = threadIdx.x & 31, tr = threadIdx.x >> 5;   // 32 x 8
#pragma unroll
    for (int i = 0; i < 4; ++i)
        tile[tr + 8 * i][tc] = src[(size_t)(r0 + tr + 8 * i) * C + (c0 + tc)];
    __syncthreads();
#pragma unroll
    for (int i = 0; i < 4; ++i)
        dst[(size_t)(c0 + tr + 8 * i) * R + (r0 + tc)] = f2bf(tile[tc][tr + 8 * i]);
}

// ---------------------------------------------------------------------------
// Small helper kernels
// ---------------------------------------------------------------------------

__global__ void k_zero16(int* counts) { if (threadIdx.x < E) counts[threadIdx.x] = 0; }

__global__ void k_cvt_bf16(const float* __restrict__ src, bf16_t* __restrict__ dst, int n) {
    int i = blockIdx.x * 256 + threadIdx.x;
    if (i < n) dst[i] = f2bf(src[i]);
}

__global__ void k_outinit(float* __restrict__ out, const float* __restrict__ sb2) {
    int i = blockIdx.x * 256 + threadIdx.x;
    if (i < NTOK * D) {
        int d = i & (D - 1);
        out[i] = sb2[d] + sb2[D + d];
    }
}

// One wave per token: lanes = (expert e = lane&15, half = lane>>4).
__global__ void __launch_bounds__(256)
k_gate(const float* __restrict__ x, const float* __restrict__ gw,
       const float* __restrict__ gb, int* __restrict__ topi,
       float* __restrict__ topw, int* __restrict__ counts)
{
    const int wid  = (blockIdx.x * blockDim.x + threadIdx.x) >> 5;   // token
    const int lane = threadIdx.x & 31;
    if (wid >= NTOK) return;
    const int e    = lane & 15;
    const int half = lane >> 4;
    const float* xr = x + (size_t)wid * D;

    float acc = 0.f;
    for (int d = half; d < D; d += 2)
        acc = fmaf(xr[d], gw[(size_t)d * E + e], acc);
    acc += __shfl_xor(acc, 16, 32);
    const float logit = acc + gb[e];

    float m = logit;
#pragma unroll
    for (int o = 8; o; o >>= 1) m = fmaxf(m, __shfl_xor(m, o, 32));
    const float ex = __expf(logit - m);
    float sum = ex;
#pragma unroll
    for (int o = 8; o; o >>= 1) sum += __shfl_xor(sum, o, 32);
    const float prob = ex / sum;

    float v1 = prob; int i1 = e;
#pragma unroll
    for (int o = 8; o; o >>= 1) {
        float ov = __shfl_xor(v1, o, 32);
        int   oi = __shfl_xor(i1, o, 32);
        if (ov > v1 || (ov == v1 && oi < i1)) { v1 = ov; i1 = oi; }
    }
    float v2 = (e == i1) ? -1.f : prob; int i2 = e;
#pragma unroll
    for (int o = 8; o; o >>= 1) {
        float ov = __shfl_xor(v2, o, 32);
        int   oi = __shfl_xor(i2, o, 32);
        if (ov > v2 || (ov == v2 && oi < i2)) { v2 = ov; i2 = oi; }
    }
    if (lane == 0) {
        topi[wid * 2]     = i1; topi[wid * 2 + 1] = i2;
        topw[wid * 2]     = v1; topw[wid * 2 + 1] = v2;
        atomicAdd(&counts[i1], 1);
        atomicAdd(&counts[i2], 1);
    }
}

__global__ void k_scan(const int* __restrict__ counts, int* __restrict__ offs,
                       int* __restrict__ fill) {
    if (threadIdx.x == 0) {
        int acc = 0;
        for (int e = 0; e < E; ++e) { offs[e] = acc; fill[e] = acc; acc += counts[e]; }
        offs[E] = acc;
    }
}

__global__ void k_scatter(const int* __restrict__ topi, const float* __restrict__ topw,
                          int* __restrict__ fill, int* __restrict__ btok,
                          float* __restrict__ bw) {
    int n = blockIdx.x * 256 + threadIdx.x;
    if (n >= NTOK) return;
#pragma unroll
    for (int j = 0; j < 2; ++j) {
        int e = topi[n * 2 + j];
        int pos = atomicAdd(&fill[e], 1);
        btok[pos] = n;
        bw[pos]   = topw[n * 2 + j];
    }
}

// ---------------------------------------------------------------------------
// GEMM kernels (64x64 tile / 256-thread block / 8 waves x 2 WMMA tiles)
// All weight operands are the pre-transposed bf16 copies ([N][K] layout).
// ---------------------------------------------------------------------------

// Routed up-projection: h = gelu(x_gather @ ew1[e] + eb1[e]) -> bf16 buffer
__global__ void __launch_bounds__(256)
k_routed_h(const bf16_t* __restrict__ xb, const bf16_t* __restrict__ ew1t,
           const float* __restrict__ eb1, const int* __restrict__ offs,
           const int* __restrict__ btok, bf16_t* __restrict__ hr)
{
    const int e    = blockIdx.z;
    const int seg0 = offs[e];
    const int cnt  = offs[e + 1] - seg0;
    const int row0 = blockIdx.y * 64;
    if (row0 >= cnt) return;
    const int ncol = blockIdx.x * 64;

    __shared__ int s_tok[64];          // indices, not pointers -> global loads
    __shared__ int s_ok[64];
    const int tid = threadIdx.x;
    if (tid < 64) {
        int r = row0 + tid;
        int tok = 0, ok = 0;
        if (r < cnt) { tok = btok[seg0 + r]; ok = 1; }
        s_tok[tid] = tok;              // clamped -> always a valid load
        s_ok[tid]  = ok;
    }
    __syncthreads();

    const int w = tid >> 5, lane = tid & 31;
    const int mi0 = w >> 2, ni = w & 3;
    const int arow = lane & 15;
    const bf16_t* a0 = xb + (size_t)s_tok[mi0 * 16 + arow] * D;
    const bf16_t* a1 = xb + (size_t)s_tok[(mi0 + 2) * 16 + arow] * D;
    const int nabs = ncol + ni * 16 + (lane & 15);
    const bf16_t* bcol = ew1t + ((size_t)e * H_R + nabs) * D;   // [H_R][D]

    v8f acc0 = {}, acc1 = {};
    wave_gemm2(a0, a1, bcol, D, acc0, acc1);

    const float bias = eb1[(size_t)e * H_R + nabs];
    const int csub = (lane >> 4) * 8;
#pragma unroll
    for (int t = 0; t < 2; ++t) {
        const int mi = mi0 + 2 * t;
        v8f& acc = t ? acc1 : acc0;
#pragma unroll
        for (int r = 0; r < 8; ++r) {
            const int rr = mi * 16 + csub + r;
            if (s_ok[rr])
                hr[(size_t)(seg0 + row0 + rr) * H_R + nabs] = f2bf(gelu_exact(acc[r] + bias));
        }
    }
}

// Routed down-projection: out[token] += weight * (h @ ew2[e] + eb2[e])
__global__ void __launch_bounds__(256)
k_routed_out(const bf16_t* __restrict__ hr, const bf16_t* __restrict__ ew2t,
             const float* __restrict__ eb2, const int* __restrict__ offs,
             const int* __restrict__ btok, const float* __restrict__ bw,
             float* __restrict__ out)
{
    const int e    = blockIdx.z;
    const int seg0 = offs[e];
    const int cnt  = offs[e + 1] - seg0;
    const int row0 = blockIdx.y * 64;
    if (row0 >= cnt) return;
    const int ncol = blockIdx.x * 64;

    __shared__ int   s_tok[64];
    __shared__ float s_w[64];
    const int tid = threadIdx.x;
    if (tid < 64) {
        int r = row0 + tid;
        if (r < cnt) { s_tok[tid] = btok[seg0 + r]; s_w[tid] = bw[seg0 + r]; }
        else         { s_tok[tid] = -1;             s_w[tid] = 0.f; }
    }
    __syncthreads();

    const int w = tid >> 5, lane = tid & 31;
    const int mi0 = w >> 2, ni = w & 3;
    const int arow = lane & 15;
    int r0 = seg0 + row0 + mi0 * 16 + arow;       if (r0 > NA - 1) r0 = NA - 1;
    int r1 = seg0 + row0 + (mi0 + 2) * 16 + arow; if (r1 > NA - 1) r1 = NA - 1;
    const bf16_t* a0 = hr + (size_t)r0 * H_R;
    const bf16_t* a1 = hr + (size_t)r1 * H_R;
    const int nabs = ncol + ni * 16 + (lane & 15);
    const bf16_t* bcol = ew2t + ((size_t)e * D + nabs) * H_R;   // [D][H_R]

    v8f acc0 = {}, acc1 = {};
    wave_gemm2(a0, a1, bcol, H_R, acc0, acc1);

    const float bias = eb2[(size_t)e * D + nabs];
    const int csub = (lane >> 4) * 8;
#pragma unroll
    for (int t = 0; t < 2; ++t) {
        const int mi = mi0 + 2 * t;
        v8f& acc = t ? acc1 : acc0;
#pragma unroll
        for (int r = 0; r < 8; ++r) {
            const int rr = mi * 16 + csub + r;
            const int tok = s_tok[rr];
            if (tok >= 0)
                atomicAdd(&out[(size_t)tok * D + nabs], (acc[r] + bias) * s_w[rr]);
        }
    }
}

// Shared up-projection (hidden chunk [c0, c0+CH)): hs = gelu(x @ sw1[s] + sb1[s])
__global__ void __launch_bounds__(256)
k_shared_h(const bf16_t* __restrict__ xb, const bf16_t* __restrict__ sw1t,
           const float* __restrict__ sb1, bf16_t* __restrict__ hs, int c0)
{
    const int s    = blockIdx.z;
    const int row0 = blockIdx.y * 64;
    const int ncol = blockIdx.x * 64;            // within chunk
    const int tid  = threadIdx.x;
    const int w = tid >> 5, lane = tid & 31;
    const int mi0 = w >> 2, ni = w & 3;
    const int arow = lane & 15;
    const bf16_t* a0 = xb + (size_t)(row0 + mi0 * 16 + arow) * D;
    const bf16_t* a1 = xb + (size_t)(row0 + (mi0 + 2) * 16 + arow) * D;
    const int nloc = ncol + ni * 16 + (lane & 15);
    const int nabs = c0 + nloc;
    const bf16_t* bcol = sw1t + ((size_t)s * H_S + nabs) * D;   // [H_S][D]

    v8f acc0 = {}, acc1 = {};
    wave_gemm2(a0, a1, bcol, D, acc0, acc1);

    const float bias = sb1[(size_t)s * H_S + nabs];
    const int csub = (lane >> 4) * 8;
#pragma unroll
    for (int t = 0; t < 2; ++t) {
        const int mi = mi0 + 2 * t;
        v8f& acc = t ? acc1 : acc0;
#pragma unroll
        for (int r = 0; r < 8; ++r) {
            const int row = row0 + mi * 16 + csub + r;
            hs[((size_t)s * NTOK + row) * CH + nloc] = f2bf(gelu_exact(acc[r] + bias));
        }
    }
}

// Shared down-projection (chunk accumulate): out += hs @ sw2[s][c0:c0+CH, :]
__global__ void __launch_bounds__(256)
k_shared_out(const bf16_t* __restrict__ hs, const bf16_t* __restrict__ sw2t,
             float* __restrict__ out, int c0)
{
    const int s    = blockIdx.z;
    const int row0 = blockIdx.y * 64;
    const int ncol = blockIdx.x * 64;
    const int tid  = threadIdx.x;
    const int w = tid >> 5, lane = tid & 31;
    const int mi0 = w >> 2, ni = w & 3;
    const int arow = lane & 15;
    const bf16_t* a0 = hs + ((size_t)s * NTOK + row0 + mi0 * 16 + arow) * CH;
    const bf16_t* a1 = hs + ((size_t)s * NTOK + row0 + (mi0 + 2) * 16 + arow) * CH;
    const int nabs = ncol + ni * 16 + (lane & 15);
    // sw2t layout [D][H_S]; K-window [c0, c0+CH)
    const bf16_t* bcol = sw2t + ((size_t)s * D + nabs) * H_S + c0;

    v8f acc0 = {}, acc1 = {};
    wave_gemm2(a0, a1, bcol, CH, acc0, acc1);

    const int csub = (lane >> 4) * 8;
#pragma unroll
    for (int t = 0; t < 2; ++t) {
        const int mi = mi0 + 2 * t;
        v8f& acc = t ? acc1 : acc0;
#pragma unroll
        for (int r = 0; r < 8; ++r) {
            const int row = row0 + mi * 16 + csub + r;
            atomicAdd(&out[(size_t)row * D + nabs], acc[r]);
        }
    }
}

// ---------------------------------------------------------------------------
// Launch
// ---------------------------------------------------------------------------

extern "C" void kernel_launch(void* const* d_in, const int* in_sizes, int n_in,
                              void* d_out, int out_size, void* d_ws, size_t ws_size,
                              hipStream_t stream)
{
    (void)in_sizes; (void)n_in; (void)out_size; (void)ws_size;
    const float* x   = (const float*)d_in[0];
    const float* gw  = (const float*)d_in[1];
    const float* gb  = (const float*)d_in[2];
    const float* ew1 = (const float*)d_in[3];
    const float* eb1 = (const float*)d_in[4];
    const float* ew2 = (const float*)d_in[5];
    const float* eb2 = (const float*)d_in[6];
    const float* sw1 = (const float*)d_in[7];
    const float* sb1 = (const float*)d_in[8];
    const float* sw2 = (const float*)d_in[9];
    const float* sb2 = (const float*)d_in[10];
    float* out = (float*)d_out;

    // Workspace carve (~320 MB total)
    char* p = (char*)d_ws;
    auto take = [&](size_t bytes) -> char* {
        char* r = p; p += (bytes + 255) & ~(size_t)255; return r;
    };
    bf16_t* xb    = (bf16_t*)take((size_t)NTOK * D * sizeof(bf16_t));        // 33.5 MB
    bf16_t* hr    = (bf16_t*)take((size_t)NA * H_R * sizeof(bf16_t));        // 16.8 MB
    bf16_t* hs    = (bf16_t*)take((size_t)2 * NTOK * CH * sizeof(bf16_t));   // 67 MB
    bf16_t* ew1t  = (bf16_t*)take((size_t)E * D * H_R * sizeof(bf16_t));     // 33.5 MB
    bf16_t* ew2t  = (bf16_t*)take((size_t)E * H_R * D * sizeof(bf16_t));     // 33.5 MB
    bf16_t* sw1t  = (bf16_t*)take((size_t)2 * D * H_S * sizeof(bf16_t));     // 67 MB
    bf16_t* sw2t  = (bf16_t*)take((size_t)2 * H_S * D * sizeof(bf16_t));     // 67 MB
    int*    topi  = (int*)  take((size_t)NTOK * 2 * sizeof(int));
    float*  topw  = (float*)take((size_t)NTOK * 2 * sizeof(float));
    int*    btok  = (int*)  take((size_t)NA * sizeof(int));
    float*  bw    = (float*)take((size_t)NA * sizeof(float));
    int*    counts= (int*)  take(64 * sizeof(int));
    int*    offs  = (int*)  take(64 * sizeof(int));
    int*    fill  = (int*)  take(64 * sizeof(int));

    // Weight pre-pass: f32 [K][N] -> bf16 [N][K]
    k_transpose_cvt<<<dim3(H_R / 32, D   / 32, E), 256, 0, stream>>>(ew1, ew1t, D,   H_R);
    k_transpose_cvt<<<dim3(D   / 32, H_R / 32, E), 256, 0, stream>>>(ew2, ew2t, H_R, D);
    k_transpose_cvt<<<dim3(H_S / 32, D   / 32, 2), 256, 0, stream>>>(sw1, sw1t, D,   H_S);
    k_transpose_cvt<<<dim3(D   / 32, H_S / 32, 2), 256, 0, stream>>>(sw2, sw2t, H_S, D);

    k_zero16  <<<1, 32, 0, stream>>>(counts);
    k_cvt_bf16<<<(NTOK * D + 255) / 256, 256, 0, stream>>>(x, xb, NTOK * D);
    k_gate    <<<NTOK / 8, 256, 0, stream>>>(x, gw, gb, topi, topw, counts);
    k_scan    <<<1, 32, 0, stream>>>(counts, offs, fill);
    k_scatter <<<(NTOK + 255) / 256, 256, 0, stream>>>(topi, topw, fill, btok, bw);
    k_outinit <<<(NTOK * D + 255) / 256, 256, 0, stream>>>(out, sb2);

    k_routed_h  <<<dim3(H_R / 64, NTOK / 64, E), 256, 0, stream>>>(xb, ew1t, eb1, offs, btok, hr);
    k_routed_out<<<dim3(D   / 64, NTOK / 64, E), 256, 0, stream>>>(hr, ew2t, eb2, offs, btok, bw, out);

    for (int c = 0; c < NCH; ++c) {
        k_shared_h  <<<dim3(CH / 64, NTOK / 64, 2), 256, 0, stream>>>(xb, sw1t, sb1, hs, c * CH);
        k_shared_out<<<dim3(D  / 64, NTOK / 64, 2), 256, 0, stream>>>(hs, sw2t, out, c * CH);
    }
}
